// SOFTAttention_22445499089592
// MI455X (gfx1250) — compile-verified, matrix operand order checked by reading, not run on hardware
//
#include <hip/hip_runtime.h>
#include <hip/hip_bf16.h>

// ---------------------------------------------------------------------------
// SOFT attention (landmark Gaussian-kernel attention + Newton inverse) for
// MI455X / gfx1250.
//   - heavy GEMMs:        v_wmma_f32_16x16x32_f16 (f32 accumulate)
//   - Newton iteration:   v_wmma_f32_16x16x4_f32 (precision-critical),
//                         A-strip staged in LDS via Tensor Data Mover
// ---------------------------------------------------------------------------

typedef __attribute__((ext_vector_type(16))) _Float16     v16h;
typedef __attribute__((ext_vector_type(8)))  float        v8f;
typedef __attribute__((ext_vector_type(2)))  float        v2f;
typedef __attribute__((ext_vector_type(4)))  unsigned int v4u;
typedef __attribute__((ext_vector_type(8)))  int          v8i;
typedef __attribute__((ext_vector_type(4)))  int          v4i;

#define BATCH   16
#define NTOK    1025
#define NPATCH  1024
#define HEADS   12
#define BH      (BATCH * HEADS)   // 192
#define DH      64
#define CH      768
#define OC      2304              // 3*CH
#define LM      256               // landmarks (16x16)
#define TAU     8.0f              // sqrt(HEAD_DIM)
#define NEWTON  6

#define TDM_OK (__has_builtin(__builtin_amdgcn_tensor_load_to_lds) && \
                __has_builtin(__builtin_amdgcn_s_wait_tensorcnt))

// ---------------- WMMA wrappers -------------------------------------------

__device__ __forceinline__ v8f wmma_f16(v16h a, v16h b, v8f c) {
  return __builtin_amdgcn_wmma_f32_16x16x32_f16(false, a, false, b, (short)0, c,
                                                false, false);
}

__device__ __forceinline__ v8f wmma_f32(v2f a, v2f b, v8f c) {
  return __builtin_amdgcn_wmma_f32_16x16x4_f32(false, a, false, b, (short)0, c,
                                               false, false);
}

// 16-bit A-matrix 16x32 layout (ISA 7.12.2): per lane two contiguous 8-half
// (16B) chunks at k-offsets hi*8 and 16+hi*8.  Same loader serves B when the
// source is "N-major, K-contiguous" (W[o][k] rows, transposed V, ...).
__device__ __forceinline__ v16h load_tile_f16(const _Float16* p, int ld) {
  int lane = threadIdx.x & 31;
  int m = lane & 15, hi = lane >> 4;
  const _Float16* r = p + (size_t)m * ld;
  v16h a;
#pragma unroll
  for (int e = 0; e < 8; ++e) a[e] = r[hi * 8 + e];
#pragma unroll
  for (int e = 0; e < 8; ++e) a[8 + e] = r[16 + hi * 8 + e];
  return a;
}

// 32-bit A-matrix 16x4: lane l (m=l&15, hi=l>>4), vgpr v -> K = hi*2 + v.
__device__ __forceinline__ v2f load_a_f32(const float* p, int ld) {
  int lane = threadIdx.x & 31;
  int m = lane & 15, hi = lane >> 4;
  v2f a;
  a.x = p[(size_t)m * ld + hi * 2 + 0];
  a.y = p[(size_t)m * ld + hi * 2 + 1];
  return a;
}

// 32-bit B-matrix 4x16 from a row-major KxN source (coalesced across lanes).
__device__ __forceinline__ v2f load_b_f32(const float* p, int ld) {
  int lane = threadIdx.x & 31;
  int n = lane & 15, hi = lane >> 4;
  v2f b;
  b.x = p[(size_t)(hi * 2 + 0) * ld + n];
  b.y = p[(size_t)(hi * 2 + 1) * ld + n];
  return b;
}

// ---------------- small utility kernels -----------------------------------

__global__ void k_cvt_f32_to_f16(const float* __restrict__ src,
                                 _Float16* __restrict__ dst, size_t total) {
  size_t i = (size_t)blockIdx.x * blockDim.x + threadIdx.x;
  if (i < total) dst[i] = (_Float16)src[i];
}

__global__ void k_zero_f32(float* __restrict__ p, int n) {
  int i = blockIdx.x * blockDim.x + threadIdx.x;
  if (i < n) p[i] = 0.0f;
}

// squared-norm over the last dim (64) of an [rows,64] f16 tensor
__global__ void k_norm(const _Float16* __restrict__ t,
                       float* __restrict__ nrm, int rows) {
  int i = blockIdx.x * blockDim.x + threadIdx.x;
  if (i >= rows) return;
  const _Float16* p = t + (size_t)i * DH;
  float s = 0.0f;
  for (int d = 0; d < DH; ++d) { float x = (float)p[d]; s += x * x; }
  nrm[i] = s;
}

// 2x2 spatial mean-pool of the 32x32 patch grid -> 16x16 landmarks (+norms)
__global__ __launch_bounds__(64) void k_pool(const _Float16* __restrict__ th,
                                             _Float16* __restrict__ landh,
                                             float* __restrict__ landn) {
  int bh = blockIdx.x;          // 192
  int l  = blockIdx.y;          // 256
  int d  = threadIdx.x;         // 64
  int ly = l >> 4, lx = l & 15;
  float s = 0.0f;
#pragma unroll
  for (int dy = 0; dy < 2; ++dy)
#pragma unroll
    for (int dx = 0; dx < 2; ++dx) {
      int n = 1 + (2 * ly + dy) * 32 + (2 * lx + dx);
      s += (float)th[((size_t)bh * NTOK + n) * DH + d];
    }
  s *= 0.25f;
  landh[((size_t)bh * LM + l) * DH + d] = (_Float16)s;
  __shared__ float red[64];
  red[d] = s * s;
  __syncthreads();
  for (int off = 32; off > 0; off >>= 1) {
    if (d < off) red[d] += red[d + off];
    __syncthreads();
  }
  if (d == 0) landn[(size_t)bh * LM + l] = red[0];
}

// ---------------- QKV projection (f16 WMMA GEMM) --------------------------
// One wave computes a 16x64 strip (4 accumulators): the A fragment is loaded
// once per K-step and reused by 4 WMMAs.  Writes q,k in [bh][n][d] and v
// TRANSPOSED [bh][d][n] so it can serve as a WMMA B operand later.
__global__ __launch_bounds__(32) void k_qkv_gemm(
    const _Float16* __restrict__ xh, const _Float16* __restrict__ wh,
    const float* __restrict__ bias, _Float16* __restrict__ qh,
    _Float16* __restrict__ kh, _Float16* __restrict__ vTh) {
  int mt = blockIdx.x;   // 0..1024
  int og = blockIdx.y;   // 0..35  (four 16-col tiles each)
  size_t r0 = (size_t)mt * 16;
  v8f acc[4] = {};
  for (int k0 = 0; k0 < CH; k0 += 32) {
    v16h a = load_tile_f16(xh + r0 * CH + k0, CH);
#pragma unroll
    for (int u = 0; u < 4; ++u) {
      v16h b = load_tile_f16(wh + (size_t)(og * 4 + u) * 16 * CH + k0, CH);
      acc[u] = wmma_f16(a, b, acc[u]);
    }
  }
  int lane = threadIdx.x & 31;
  int n = lane & 15, hi = lane >> 4;
#pragma unroll
  for (int u = 0; u < 4; ++u) {
    int o = (og * 4 + u) * 16 + n;
    float bv = bias[o];
    int t = o / CH, c = o % CH, h = c / DH, d = c % DH;
#pragma unroll
    for (int v = 0; v < 8; ++v) {
      int r = (int)r0 + v + 8 * hi;
      int b_ = r / NTOK, nn = r % NTOK;
      _Float16 val = (_Float16)(acc[u][v] + bv);
      size_t bh = (size_t)b_ * HEADS + h;
      if (t == 0)      qh[(bh * NTOK + nn) * DH + d] = val;
      else if (t == 1) kh[(bh * NTOK + nn) * DH + d] = val;
      else             vTh[(bh * DH + d) * NTOK + nn] = val;
    }
  }
}

// ---------------- M2 = gauss(q_land, k_land) ------------------------------
__global__ __launch_bounds__(32) void k_m2(
    const _Float16* __restrict__ qlh, const _Float16* __restrict__ klh,
    const float* __restrict__ qln, const float* __restrict__ kln,
    float* __restrict__ M2) {
  int bh = blockIdx.x, it = blockIdx.y, jt = blockIdx.z;
  const _Float16* qa = qlh + ((size_t)bh * LM + it * 16) * DH;
  const _Float16* kb = klh + ((size_t)bh * LM + jt * 16) * DH;
  v8f acc = {};
  acc = wmma_f16(load_tile_f16(qa, DH),      load_tile_f16(kb, DH),      acc);
  acc = wmma_f16(load_tile_f16(qa + 32, DH), load_tile_f16(kb + 32, DH), acc);
  int lane = threadIdx.x & 31;
  int n = lane & 15, hi = lane >> 4;
  float knv = kln[(size_t)bh * LM + jt * 16 + n];
  float* out = M2 + (size_t)bh * LM * LM;
#pragma unroll
  for (int v = 0; v < 8; ++v) {
    int m = v + 8 * hi;
    float dist = qln[(size_t)bh * LM + it * 16 + m] + knv - 2.0f * acc[v];
    out[(size_t)(it * 16 + m) * LM + jt * 16 + n] = __expf(-dist / TAU);
  }
}

// max over rows of row-sums (entries positive -> int-bit atomicMax is valid)
__global__ __launch_bounds__(256) void k_rowmax(const float* __restrict__ M2,
                                                float* __restrict__ nrm) {
  int bh = blockIdx.x, i = blockIdx.y;
  int j = threadIdx.x;
  __shared__ float red[256];
  red[j] = M2[((size_t)bh * LM + i) * LM + j];
  __syncthreads();
  for (int off = 128; off > 0; off >>= 1) {
    if (j < off) red[j] += red[j + off];
    __syncthreads();
  }
  if (j == 0) atomicMax((int*)&nrm[bh], __float_as_int(red[0]));
}

// inv0 = M2^T / (norm^2 + 1e-6)
__global__ void k_inv0(const float* __restrict__ M2,
                       const float* __restrict__ nrm,
                       float* __restrict__ inv) {
  size_t idx = (size_t)blockIdx.x * blockDim.x + threadIdx.x;
  if (idx >= (size_t)BH * LM * LM) return;
  int bh  = (int)(idx / (LM * LM));
  int rem = (int)(idx % (LM * LM));
  int i = rem / LM, j = rem % LM;
  float nv = nrm[bh];
  inv[idx] = M2[((size_t)bh * LM + j) * LM + i] / (nv * nv + 1e-6f);
}

// ---------------- Newton step GEMM (f32 WMMA + TDM-staged LDS) ------------
// Workgroup = 4 waves computes a 16x256 strip of Out for one (bh, it).
// The 16x256 f32 A-strip is DMA'd into LDS by the Tensor Data Mover once and
// reused by all waves; each wave keeps 4 accumulators (16x64) so the LDS A
// fragment is reused 4x per K-step.   diag2: Out = 2I - A@B, else A@B.
__global__ __launch_bounds__(128) void k_newton(const float* __restrict__ A,
                                                const float* __restrict__ Bm,
                                                float* __restrict__ Out,
                                                int diag2) {
  int bh = blockIdx.x, it = blockIdx.y;   // 192 x 16
  __shared__ float sA[16 * LM];           // 16 KB strip, LDS offset 0
  const float* Asrc = A + (size_t)bh * LM * LM + (size_t)it * 16 * LM;
  int tid = threadIdx.x;
  int wave = tid >> 5;
  int lane = tid & 31;

#if TDM_OK
  if (wave == 0) {
    // Tensor DMA descriptor (CDNA5 ISA): count=1, type=2("image"),
    // data_size=4B, 2D tensor 256x256, tile = 256(x) x 16(y), stride0=256.
    unsigned long long ga = (unsigned long long)(uintptr_t)Asrc;
    v4u g0;
    g0[0] = 1u;                                   // count=1, user mode
    g0[1] = 0u;                                   // lds_addr = 0 (sA)
    g0[2] = (unsigned)ga;                         // global_addr[31:0]
    g0[3] = (unsigned)((ga >> 32) & 0x1FFFFFFu)   // global_addr[56:32]
            | (2u << 30);                         // type=2
    v8i g1;
    g1[0] = (int)(2u << 16);                      // data_size=2 (4 bytes)
    g1[1] = (int)((unsigned)LM << 16);            // tensor_dim0[15:0]
    g1[2] = (int)((unsigned)LM << 16);            // dim0[31:16]=0 | dim1[15:0]
    g1[3] = (int)((unsigned)LM << 16);            // dim1[31:16]=0 | tile_dim0
    g1[4] = 16;                                   // tile_dim1=16, tile_dim2=0
    g1[5] = LM;                                   // tensor_dim0_stride[31:0]
    g1[6] = 0;
    g1[7] = 0;
    v4i z4 = {};
#if defined(__clang_major__) && (__clang_major__ >= 23)
    v8i z8 = {};
    __builtin_amdgcn_tensor_load_to_lds(g0, g1, z4, z4, z8, 0);
#else
    __builtin_amdgcn_tensor_load_to_lds(g0, g1, z4, z4, 0);
#endif
    __builtin_amdgcn_s_wait_tensorcnt(0);
  }
  __syncthreads();
#else
  // fallback: cooperative staged copy (strip is contiguous 4096 floats)
  for (int idx = tid; idx < 16 * LM / 4; idx += 128)
    ((float4*)sA)[idx] = ((const float4*)Asrc)[idx];
  __syncthreads();
#endif

  const float* Bb = Bm + (size_t)bh * LM * LM + (size_t)wave * 64;
  v8f acc[4] = {};
  for (int k0 = 0; k0 < LM; k0 += 4) {
    v2f a = load_a_f32(sA + k0, LM);               // from LDS
#pragma unroll
    for (int u = 0; u < 4; ++u)
      acc[u] = wmma_f32(a, load_b_f32(Bb + (size_t)k0 * LM + u * 16, LM),
                        acc[u]);
  }
  int n = lane & 15, hi = lane >> 4;
  float* out = Out + (size_t)bh * LM * LM;
#pragma unroll
  for (int u = 0; u < 4; ++u)
#pragma unroll
    for (int v = 0; v < 8; ++v) {
      int i = it * 16 + v + 8 * hi;
      int j = wave * 64 + u * 16 + n;
      float val = acc[u][v];
      if (diag2) val = (i == j ? 2.0f : 0.0f) - val;
      out[(size_t)i * LM + j] = val;
    }
}

// ---------------- fused  Av = M3 @ v_patch  -------------------------------
// S-tile (gauss kernel) via f16 WMMA -> exp -> LDS relayout -> second WMMA.
__global__ __launch_bounds__(32) void k_m3v(
    const _Float16* __restrict__ qlh, const _Float16* __restrict__ kh,
    const _Float16* __restrict__ vTh, const float* __restrict__ qln,
    const float* __restrict__ kn, float* __restrict__ Av) {
  int bh = blockIdx.x, it = blockIdx.y;   // 192 x 16
  __shared__ _Float16 sS[16 * 32];
  const _Float16* qa = qlh + ((size_t)bh * LM + it * 16) * DH;
  int lane = threadIdx.x & 31;
  int n = lane & 15, hi = lane >> 4;
  v8f acc[4] = {};
  v16h aq0 = load_tile_f16(qa, DH);
  v16h aq1 = load_tile_f16(qa + 32, DH);
  float qn_m[8];
#pragma unroll
  for (int v = 0; v < 8; ++v)
    qn_m[v] = qln[(size_t)bh * LM + it * 16 + v + 8 * hi];

  for (int jc = 0; jc < NPATCH; jc += 32) {
#pragma unroll
    for (int half = 0; half < 2; ++half) {
      int j0 = jc + half * 16;
      const _Float16* kb = kh + ((size_t)bh * NTOK + 1 + j0) * DH;
      v8f s = {};
      s = wmma_f16(aq0, load_tile_f16(kb, DH), s);
      s = wmma_f16(aq1, load_tile_f16(kb + 32, DH), s);
      float knv = kn[(size_t)bh * NTOK + 1 + j0 + n];
#pragma unroll
      for (int v = 0; v < 8; ++v) {
        float dist = qn_m[v] + knv - 2.0f * s[v];
        sS[(v + 8 * hi) * 32 + half * 16 + n] = (_Float16)__expf(-dist / TAU);
      }
    }
    __syncthreads();
    v16h a2 = load_tile_f16(sS, 32);
    __syncthreads();
#pragma unroll
    for (int dt = 0; dt < 4; ++dt) {
      const _Float16* vb = vTh + ((size_t)bh * DH + dt * 16) * NTOK + 1 + jc;
      acc[dt] = wmma_f16(a2, load_tile_f16(vb, NTOK), acc[dt]);
    }
  }
  float* out = Av + ((size_t)bh * LM + it * 16) * DH;
#pragma unroll
  for (int dt = 0; dt < 4; ++dt)
#pragma unroll
    for (int v = 0; v < 8; ++v)
      out[(size_t)(v + 8 * hi) * DH + dt * 16 + n] = acc[dt][v];
}

// ---------------- V_mixed = inv @ Av  (f32 WMMA), stored transposed f16 ---
__global__ __launch_bounds__(32) void k_vmixed(const float* __restrict__ inv,
                                               const float* __restrict__ Av,
                                               _Float16* __restrict__ VmT) {
  int bh = blockIdx.x, it = blockIdx.y, dt = blockIdx.z;  // 192 x 16 x 4
  const float* Ab = inv + (size_t)bh * LM * LM + (size_t)it * 16 * LM;
  const float* Bb = Av + (size_t)bh * LM * DH + dt * 16;
  v8f acc = {};
  for (int k0 = 0; k0 < LM; k0 += 4)
    acc = wmma_f32(load_a_f32(Ab + k0, LM),
                   load_b_f32(Bb + (size_t)k0 * DH, DH), acc);
  int lane = threadIdx.x & 31;
  int n = lane & 15, hi = lane >> 4;
  _Float16* out = VmT + (size_t)bh * DH * LM;
#pragma unroll
  for (int v = 0; v < 8; ++v) {
    int i = it * 16 + v + 8 * hi, d = dt * 16 + n;
    out[(size_t)d * LM + i] = (_Float16)acc[v];
  }
}

// ---------------- fused  y_patch = M1 @ V_mixed ---------------------------
__global__ __launch_bounds__(32) void k_ypatch(
    const _Float16* __restrict__ qh, const _Float16* __restrict__ klh,
    const _Float16* __restrict__ VmTh, const float* __restrict__ qn,
    const float* __restrict__ kln, _Float16* __restrict__ yh) {
  int bh = blockIdx.x, it = blockIdx.y;   // 192 x 64
  int b = bh / HEADS, h = bh % HEADS;
  __shared__ _Float16 sS[16 * 32];
  const _Float16* qa = qh + ((size_t)bh * NTOK + 1 + it * 16) * DH;
  int lane = threadIdx.x & 31;
  int n = lane & 15, hi = lane >> 4;
  v8f acc[4] = {};
  v16h aq0 = load_tile_f16(qa, DH);
  v16h aq1 = load_tile_f16(qa + 32, DH);
  float qn_m[8];
#pragma unroll
  for (int v = 0; v < 8; ++v)
    qn_m[v] = qn[(size_t)bh * NTOK + 1 + it * 16 + v + 8 * hi];

  for (int jc = 0; jc < LM; jc += 32) {
#pragma unroll
    for (int half = 0; half < 2; ++half) {
      int j0 = jc + half * 16;
      const _Float16* kb = klh + ((size_t)bh * LM + j0) * DH;
      v8f s = {};
      s = wmma_f16(aq0, load_tile_f16(kb, DH), s);
      s = wmma_f16(aq1, load_tile_f16(kb + 32, DH), s);
      float knv = kln[(size_t)bh * LM + j0 + n];
#pragma unroll
      for (int v = 0; v < 8; ++v) {
        float dist = qn_m[v] + knv - 2.0f * s[v];
        sS[(v + 8 * hi) * 32 + half * 16 + n] = (_Float16)__expf(-dist / TAU);
      }
    }
    __syncthreads();
    v16h a2 = load_tile_f16(sS, 32);
    __syncthreads();
#pragma unroll
    for (int dt = 0; dt < 4; ++dt) {
      const _Float16* vb = VmTh + ((size_t)bh * DH + dt * 16) * LM + jc;
      acc[dt] = wmma_f16(a2, load_tile_f16(vb, LM), acc[dt]);
    }
  }
#pragma unroll
  for (int dt = 0; dt < 4; ++dt)
#pragma unroll
    for (int v = 0; v < 8; ++v) {
      int i = it * 16 + v + 8 * hi;
      int d = dt * 16 + n;
      yh[((size_t)b * NTOK + 1 + i) * CH + h * DH + d] = (_Float16)acc[dt][v];
    }
}

// ---------------- y_cls (1 row per bh) ------------------------------------
__global__ __launch_bounds__(64) void k_ycls(
    const _Float16* __restrict__ qh, const _Float16* __restrict__ klh,
    const _Float16* __restrict__ VmTh, const float* __restrict__ qn,
    const float* __restrict__ kln, _Float16* __restrict__ yh) {
  int bh = blockIdx.x;
  int b = bh / HEADS, h = bh % HEADS;
  int t = threadIdx.x;   // 64
  __shared__ float w[LM];
  const _Float16* qc = qh + (size_t)bh * NTOK * DH;   // token 0
  float qcn = qn[(size_t)bh * NTOK];
  for (int j = t; j < LM; j += 64) {
    const _Float16* kb = klh + ((size_t)bh * LM + j) * DH;
    float dot = 0.0f;
    for (int d = 0; d < DH; ++d) dot += (float)qc[d] * (float)kb[d];
    float dist = qcn + kln[(size_t)bh * LM + j] - 2.0f * dot;
    w[j] = __expf(-dist / TAU);
  }
  __syncthreads();
  float acc = 0.0f;
  const _Float16* vrow = VmTh + ((size_t)bh * DH + t) * LM;
  for (int j = 0; j < LM; ++j) acc += w[j] * (float)vrow[j];
  yh[(size_t)b * NTOK * CH + h * DH + t] = (_Float16)acc;
}

// ---------------- output projection (f16 WMMA GEMM, f32 out) --------------
__global__ __launch_bounds__(32) void k_proj(const _Float16* __restrict__ yh,
                                             const _Float16* __restrict__ wh,
                                             const float* __restrict__ bias,
                                             float* __restrict__ out) {
  int mt = blockIdx.x;   // 1025
  int og = blockIdx.y;   // 0..11 (four 16-col tiles each)
  v8f acc[4] = {};
  for (int k0 = 0; k0 < CH; k0 += 32) {
    v16h a = load_tile_f16(yh + (size_t)mt * 16 * CH + k0, CH);
#pragma unroll
    for (int u = 0; u < 4; ++u)
      acc[u] = wmma_f16(a,
          load_tile_f16(wh + (size_t)(og * 4 + u) * 16 * CH + k0, CH), acc[u]);
  }
  int lane = threadIdx.x & 31;
  int n = lane & 15, hi = lane >> 4;
#pragma unroll
  for (int u = 0; u < 4; ++u) {
    int o = (og * 4 + u) * 16 + n;
    float bv = bias[o];
#pragma unroll
    for (int v = 0; v < 8; ++v) {
      size_t r = (size_t)mt * 16 + v + 8 * hi;
      out[r * CH + o] = acc[u][v] + bv;
    }
  }
}

// ---------------------------------------------------------------------------
extern "C" void kernel_launch(void* const* d_in, const int* in_sizes, int n_in,
                              void* d_out, int out_size, void* d_ws,
                              size_t ws_size, hipStream_t stream) {
  const float* x      = (const float*)d_in[0];
  const float* qkv_w  = (const float*)d_in[1];
  const float* qkv_b  = (const float*)d_in[2];
  const float* proj_w = (const float*)d_in[3];
  const float* proj_b = (const float*)d_in[4];
  float* out = (float*)d_out;

  char* ws = (char*)d_ws;
  size_t off = 0;
  auto alloc = [&](size_t bytes) -> void* {
    off = (off + 255) & ~(size_t)255;
    void* p = ws + off;
    off += bytes;
    return p;
  };

  const size_t ROWS = (size_t)BATCH * NTOK;        // 16400
  _Float16* xh    = (_Float16*)alloc(ROWS * CH * 2);
  _Float16* wqkvh = (_Float16*)alloc((size_t)OC * CH * 2);
  _Float16* wprjh = (_Float16*)alloc((size_t)CH * CH * 2);
  _Float16* qh    = (_Float16*)alloc((size_t)BH * NTOK * DH * 2);
  _Float16* kh    = (_Float16*)alloc((size_t)BH * NTOK * DH * 2);
  _Float16* vTh   = (_Float16*)alloc((size_t)BH * DH * NTOK * 2);
  float*    qn    = (float*)alloc((size_t)BH * NTOK * 4);
  float*    kn    = (float*)alloc((size_t)BH * NTOK * 4);
  _Float16* qlh   = (_Float16*)alloc((size_t)BH * LM * DH * 2);
  _Float16* klh   = (_Float16*)alloc((size_t)BH * LM * DH * 2);
  float*    qln   = (float*)alloc((size_t)BH * LM * 4);
  float*    kln   = (float*)alloc((size_t)BH * LM * 4);
  float*    M2    = (float*)alloc((size_t)BH * LM * LM * 4);
  float*    invA  = (float*)alloc((size_t)BH * LM * LM * 4);
  float*    invB  = (float*)alloc((size_t)BH * LM * LM * 4);
  float*    U     = (float*)alloc((size_t)BH * LM * LM * 4);
  float*    nrm   = (float*)alloc((size_t)BH * 4);
  float*    Av    = (float*)alloc((size_t)BH * LM * DH * 4);
  _Float16* VmTh  = (_Float16*)alloc((size_t)BH * DH * LM * 2);
  _Float16* yhh   = (_Float16*)alloc(ROWS * CH * 2);
  (void)ws_size; (void)in_sizes; (void)n_in; (void)out_size;

  // 1) f32 -> f16 conversions
  {
    size_t t = ROWS * CH;
    k_cvt_f32_to_f16<<<dim3((unsigned)((t + 255) / 256)), 256, 0, stream>>>(x, xh, t);
    t = (size_t)OC * CH;
    k_cvt_f32_to_f16<<<dim3((unsigned)((t + 255) / 256)), 256, 0, stream>>>(qkv_w, wqkvh, t);
    t = (size_t)CH * CH;
    k_cvt_f32_to_f16<<<dim3((unsigned)((t + 255) / 256)), 256, 0, stream>>>(proj_w, wprjh, t);
  }

  // 2) QKV projection (WMMA f16, 16x64 strip per wave)
  k_qkv_gemm<<<dim3(1025, 36), 32, 0, stream>>>(xh, wqkvh, qkv_b, qh, kh, vTh);

  // 3) norms for q,k (per token)
  {
    int rows = BH * NTOK;
    k_norm<<<dim3((rows + 127) / 128), 128, 0, stream>>>(qh, qn, rows);
    k_norm<<<dim3((rows + 127) / 128), 128, 0, stream>>>(kh, kn, rows);
  }

  // 4) landmarks (2x2 pool) + their norms
  k_pool<<<dim3(BH, LM), 64, 0, stream>>>(qh, qlh, qln);
  k_pool<<<dim3(BH, LM), 64, 0, stream>>>(kh, klh, kln);

  // 5) M2 and Newton seed
  k_m2<<<dim3(BH, 16, 16), 32, 0, stream>>>(qlh, klh, qln, kln, M2);
  k_zero_f32<<<1, BH, 0, stream>>>(nrm, BH);
  k_rowmax<<<dim3(BH, LM), 256, 0, stream>>>(M2, nrm);
  {
    size_t t = (size_t)BH * LM * LM;
    k_inv0<<<dim3((unsigned)((t + 255) / 256)), 256, 0, stream>>>(M2, nrm, invA);
  }

  // 6) Newton iterations (f32 WMMA + TDM LDS staging), ping-pong invA/invB
  float* cur = invA;
  float* nxt = invB;
  for (int it = 0; it < NEWTON; ++it) {
    k_newton<<<dim3(BH, 16), 128, 0, stream>>>(M2, cur, U, 1);   // U = 2I - M2@inv
    k_newton<<<dim3(BH, 16), 128, 0, stream>>>(cur, U, nxt, 0);  // inv = inv@U
    float* t = cur; cur = nxt; nxt = t;
  }

  // 7) Av = M3 @ v_patch  (fused gauss-kernel + GEMM)
  k_m3v<<<dim3(BH, 16), 32, 0, stream>>>(qlh, kh, vTh, qln, kn, Av);

  // 8) V_mixed = inv @ Av  (f32 WMMA), stored transposed f16
  k_vmixed<<<dim3(BH, 16, 4), 32, 0, stream>>>(cur, Av, VmTh);

  // 9) y_patch = M1 @ V_mixed (fused), y_cls
  k_ypatch<<<dim3(BH, 64), 32, 0, stream>>>(qh, klh, VmTh, qn, kln, yhh);
  k_ycls<<<dim3(BH), 64, 0, stream>>>(qh, klh, VmTh, qn, kln, yhh);

  // 10) output projection
  k_proj<<<dim3(1025, 12), 32, 0, stream>>>(yhh, wprjh, proj_b, out);
}
